// PathConditionalSigGenerator_28664611733737
// MI455X (gfx1250) — compile-verified, matrix operand order checked by reading, not run on hardware
//
#include <hip/hip_runtime.h>
#include <cstddef>
#include <utility>

typedef __attribute__((ext_vector_type(16))) __bf16 v16bf;
typedef __attribute__((ext_vector_type(8)))  float  v8f;

union FragBF {
    uint4 u[2];
    v16bf v;
};

struct Frags {
    FragBF a;
    FragBF b[4];
};

__device__ __forceinline__ unsigned short f2bf(float x) {
    union { float f; unsigned u; } c;
    c.f = x;
    unsigned r = c.u + 0x7FFFu + ((c.u >> 16) & 1u);
    return (unsigned short)(r >> 16);
}

// ---------------------------------------------------------------------------
// Path signature (ORDER=4, CHANNELS=2) via Chen's relation. 32 paths, one
// thread each, Lp-1 = 63 increments, 30-float signature kept in registers.
// ---------------------------------------------------------------------------
__device__ __forceinline__ void seg_exp2(float d0, float d1,
                                         float* L1, float* L2, float* L3, float* L4) {
    float dx[2] = {d0, d1};
    L1[0] = d0; L1[1] = d1;
    for (int i = 0; i < 2; ++i)
        for (int j = 0; j < 2; ++j)
            L2[i * 2 + j] = L1[i] * dx[j] * 0.5f;
    for (int p = 0; p < 4; ++p)
        for (int j = 0; j < 2; ++j)
            L3[p * 2 + j] = L2[p] * dx[j] * (1.0f / 3.0f);
    for (int p = 0; p < 8; ++p)
        for (int j = 0; j < 2; ++j)
            L4[p * 2 + j] = L3[p] * dx[j] * 0.25f;
}

__global__ void sig_kernel(const float* __restrict__ paths, float* __restrict__ sig) {
    int b = threadIdx.x;
    if (b >= 32) return;
    const float* P = paths + (size_t)b * 64 * 2;
    float S1[2], S2[4], S3[8], S4[16];
    float T1[2], T2[4], T3[8], T4[16];
    float N2[4], N3[8], N4[16];

    seg_exp2(P[2] - P[0], P[3] - P[1], S1, S2, S3, S4);

    for (int i = 1; i < 63; ++i) {
        float d0 = P[(i + 1) * 2 + 0] - P[i * 2 + 0];
        float d1 = P[(i + 1) * 2 + 1] - P[i * 2 + 1];
        seg_exp2(d0, d1, T1, T2, T3, T4);
        // chen(S, T)
        for (int p = 0; p < 4; ++p)  N2[p] = S2[p] + T2[p];
        for (int p = 0; p < 8; ++p)  N3[p] = S3[p] + T3[p];
        for (int p = 0; p < 16; ++p) N4[p] = S4[p] + T4[p];
        for (int a = 0; a < 2; ++a)
            for (int c = 0; c < 2; ++c)
                N2[a * 2 + c] += S1[a] * T1[c];
        for (int a = 0; a < 2; ++a)
            for (int c = 0; c < 4; ++c)
                N3[a * 4 + c] += S1[a] * T2[c];
        for (int a = 0; a < 4; ++a)
            for (int c = 0; c < 2; ++c)
                N3[a * 2 + c] += S2[a] * T1[c];
        for (int a = 0; a < 2; ++a)
            for (int c = 0; c < 8; ++c)
                N4[a * 8 + c] += S1[a] * T3[c];
        for (int a = 0; a < 4; ++a)
            for (int c = 0; c < 4; ++c)
                N4[a * 4 + c] += S2[a] * T2[c];
        for (int a = 0; a < 8; ++a)
            for (int c = 0; c < 2; ++c)
                N4[a * 2 + c] += S3[a] * T1[c];
        for (int a = 0; a < 2; ++a)  S1[a] += T1[a];
        for (int p = 0; p < 4; ++p)  S2[p] = N2[p];
        for (int p = 0; p < 8; ++p)  S3[p] = N3[p];
        for (int p = 0; p < 16; ++p) S4[p] = N4[p];
    }
    float* o = sig + (size_t)b * 30;   // SIG_SCALE == 1.0 so no rescale needed
    for (int p = 0; p < 2; ++p)  o[p] = S1[p];
    for (int p = 0; p < 4; ++p)  o[2 + p] = S2[p];
    for (int p = 0; p < 8; ++p)  o[6 + p] = S3[p];
    for (int p = 0; p < 16; ++p) o[14 + p] = S4[p];
}

__global__ void c0dot_kernel(const float* __restrict__ sig, const float* __restrict__ Wr,
                             float* __restrict__ c0dot) {
    int b = threadIdx.x;
    if (b >= 32) return;
    float acc = 0.0f;
    for (int j = 0; j < 30; ++j) acc += sig[b * 30 + j] * Wr[128 + j];
    c0dot[b] = acc;
}

// ---------------------------------------------------------------------------
// Pack f32 row-major W[K,N] into WMMA-B bf16 tile layout.
// Tile (kt, nt) = 32x16 bf16, stored as 256 dwords: [lane 0..31][dword 0..7].
//   k = kt*32 + 16*(d>>2) + 8*(lane>>4) + 2*(d&3)  (pair k, k+1 per dword)
// Tile order: tileIdx = (ntOffset + ntLocal) * kTilesTotal + kt.
// ---------------------------------------------------------------------------
__global__ void pack_b_kernel(const float* __restrict__ W, int K, int N,
                              unsigned int* __restrict__ dst, int kTilesTotal,
                              int ntOffset) {
    int tid = blockIdx.x * blockDim.x + threadIdx.x;
    int total = kTilesTotal * (N >> 4) * 256;
    if (tid >= total) return;
    int r = tid & 255;
    int t = tid >> 8;
    int kt  = t % kTilesTotal;
    int ntL = t / kTilesTotal;
    int lane = r >> 3;
    int d    = r & 7;
    int n = ntL * 16 + (lane & 15);
    int k = kt * 32 + ((d >> 2) << 4) + ((lane >> 4) << 3) + ((d & 3) << 1);
    unsigned lo = (k     < K) ? (unsigned)f2bf(W[(size_t)k * N + n])       : 0u;
    unsigned hi = (k + 1 < K) ? (unsigned)f2bf(W[(size_t)(k + 1) * N + n]) : 0u;
    dst[((size_t)(ntOffset + ntL) * kTilesTotal + kt) * 256 + r] = (hi << 16) | lo;
}

// ---------------------------------------------------------------------------
// Fragment loader: A fragment (2 x b128) + 4 B fragments (2 x b128 each).
// ---------------------------------------------------------------------------
template <int KT>
__device__ __forceinline__ void load_frags(Frags& F, const unsigned short* aRow,
                                           const uint4* __restrict__ Bpack,
                                           int ntb, int lane, int kt) {
    const uint4* ap = reinterpret_cast<const uint4*>(aRow + kt * 32);
    F.a.u[0] = ap[0];
    F.a.u[1] = ap[2];   // +16 bf16 elements
#pragma unroll
    for (int j = 0; j < 4; ++j) {
        const uint4* bp = Bpack + ((size_t)(ntb + j) * KT + kt) * 64 + lane * 2;
        F.b[j].u[0] = bp[0];
        F.b[j].u[1] = bp[1];
    }
}

// ---------------------------------------------------------------------------
// WMMA GEMM, bf16 in / f32 accumulate. One wave -> 16 rows x 64 cols.
// Fully unrolled K loop with double-buffered fragments so next-tile loads
// overlap the current 4-WMMA burst. Epilogue: +bias, lipswish, store bf16.
// ---------------------------------------------------------------------------
template <int KT>
__global__ __launch_bounds__(128) void gemm_lipswish(
    const unsigned short* __restrict__ A, int aStride, int aOffHi,
    const uint4* __restrict__ Bpack, int nTiles,
    const float* __restrict__ bias0, const float* __restrict__ bias1, int nSplit,
    unsigned short* __restrict__ Out, int outStride) {
    int wid  = (blockIdx.x * blockDim.x + threadIdx.x) >> 5;
    int lane = threadIdx.x & 31;
    int wpr  = nTiles >> 2;
    int mt   = wid / wpr;
    int ntb  = (wid % wpr) << 2;
    int mA   = mt * 16 + (lane & 15);
    int klo  = (lane >> 4) << 3;
    int aOff = (ntb * 16 >= nSplit) ? aOffHi : 0;
    const unsigned short* aRow = A + (size_t)mA * aStride + aOff + klo;

    v8f acc[4] = {};
    Frags frag[2];
    load_frags<KT>(frag[0], aRow, Bpack, ntb, lane, 0);
#pragma unroll
    for (int kt = 0; kt < KT; ++kt) {
        if (kt + 1 < KT)
            load_frags<KT>(frag[(kt + 1) & 1], aRow, Bpack, ntb, lane, kt + 1);
        Frags& F = frag[kt & 1];
#pragma unroll
        for (int j = 0; j < 4; ++j)
            acc[j] = __builtin_amdgcn_wmma_f32_16x16x32_bf16(
                false, F.a.v, false, F.b[j].v, (short)0, acc[j], false, false);
    }
    int rowBase = mt * 16 + ((lane >> 4) << 3);
    int col0 = lane & 15;
#pragma unroll
    for (int j = 0; j < 4; ++j) {
        int n = (ntb + j) * 16 + col0;
        float bs = (n < nSplit) ? bias0[n] : bias1[n - nSplit];
#pragma unroll
        for (int r = 0; r < 8; ++r) {
            float v = acc[j][r] + bs;
            float y = 0.909f * v / (1.0f + __expf(-v));
            Out[(size_t)(rowBase + r) * outStride + n] = f2bf(y);
        }
    }
}

// Layer-3 variant: tanh epilogue, split f32 outputs (drift 128 cols, diff 512).
template <int KT>
__global__ __launch_bounds__(128) void gemm_l3_tanh(
    const unsigned short* __restrict__ A, int aStride, int aOffHi,
    const uint4* __restrict__ Bpack, int nTiles,
    const float* __restrict__ bias0, const float* __restrict__ bias1, int nSplit,
    float* __restrict__ Fout, float* __restrict__ Gout) {
    int wid  = (blockIdx.x * blockDim.x + threadIdx.x) >> 5;
    int lane = threadIdx.x & 31;
    int wpr  = nTiles >> 2;
    int mt   = wid / wpr;
    int ntb  = (wid % wpr) << 2;
    int mA   = mt * 16 + (lane & 15);
    int klo  = (lane >> 4) << 3;
    int aOff = (ntb * 16 >= nSplit) ? aOffHi : 0;
    const unsigned short* aRow = A + (size_t)mA * aStride + aOff + klo;

    v8f acc[4] = {};
    Frags frag[2];
    load_frags<KT>(frag[0], aRow, Bpack, ntb, lane, 0);
#pragma unroll
    for (int kt = 0; kt < KT; ++kt) {
        if (kt + 1 < KT)
            load_frags<KT>(frag[(kt + 1) & 1], aRow, Bpack, ntb, lane, kt + 1);
        Frags& F = frag[kt & 1];
#pragma unroll
        for (int j = 0; j < 4; ++j)
            acc[j] = __builtin_amdgcn_wmma_f32_16x16x32_bf16(
                false, F.a.v, false, F.b[j].v, (short)0, acc[j], false, false);
    }
    int rowBase = mt * 16 + ((lane >> 4) << 3);
    int col0 = lane & 15;
#pragma unroll
    for (int j = 0; j < 4; ++j) {
        int n = (ntb + j) * 16 + col0;
        float bs = (n < nSplit) ? bias0[n] : bias1[n - nSplit];
#pragma unroll
        for (int r = 0; r < 8; ++r) {
            float v = tanhf(acc[j][r] + bs);   // TSCALE == 1.0
            int m = rowBase + r;
            if (n < nSplit) Fout[(size_t)m * 128 + n] = v;
            else            Gout[(size_t)m * 512 + (n - nSplit)] = v;
        }
    }
}

// ---------------------------------------------------------------------------
// h0 = [xT, c0] @ Wi + bi ; also initializes z, zh, and the bf16 tx buffer.
// grid: 1024 blocks (rows m), 160 threads (tx columns).
// ---------------------------------------------------------------------------
__global__ void h0_kernel(const float* __restrict__ paths, const float* __restrict__ sig,
                          const float* __restrict__ Wi, const float* __restrict__ bi,
                          const float* __restrict__ ts,
                          float* __restrict__ z, float* __restrict__ zh,
                          unsigned short* __restrict__ tx) {
    int m = blockIdx.x;
    int b = m >> 5;
    int tid = threadIdx.x;
    unsigned short* txr = tx + (size_t)m * 160;
    if (tid == 0) {
        txr[0] = f2bf(ts[0]);
    } else if (tid <= 128) {
        int h = tid - 1;
        float xT = paths[(size_t)b * 128 + 127];   // paths[b, -1, 1]
        float acc = bi[h] + xT * Wi[h];
        const float* sb = sig + (size_t)b * 30;
        for (int j = 0; j < 30; ++j) acc += sb[j] * Wi[(size_t)(1 + j) * 128 + h];
        size_t idx = (size_t)m * 128 + h;
        z[idx] = acc;
        zh[idx] = acc;
        txr[1 + h] = f2bf(acc);
    } else if (tid < 159) {
        txr[tid] = f2bf(sig[(size_t)b * 30 + (tid - 129)]);
    } else {
        txr[159] = 0;
    }
}

// ---------------------------------------------------------------------------
// u1: zh1 = 2z - zh + f*dt + g.dW ; rebuild tx for the next MLP eval.
// ---------------------------------------------------------------------------
__global__ void u1_kernel(float* __restrict__ z, float* __restrict__ zh,
                          const float* __restrict__ f, const float* __restrict__ g,
                          const float* __restrict__ dW, const float* __restrict__ sig,
                          const float* __restrict__ ts, int s, float dt,
                          unsigned short* __restrict__ tx) {
    int m = blockIdx.x;
    int b = m >> 5;
    int tid = threadIdx.x;
    unsigned short* txr = tx + (size_t)m * 160;
    if (tid == 0) {
        txr[0] = f2bf(ts[s]);
    } else if (tid <= 128) {
        int h = tid - 1;
        size_t idx = (size_t)m * 128 + h;
        const float* gr = g + (size_t)m * 512 + (size_t)h * 4;
        const float* dw = dW + (size_t)m * 4;
        float acc = 2.0f * z[idx] - zh[idx] + f[idx] * dt;
        acc += gr[0] * dw[0] + gr[1] * dw[1] + gr[2] * dw[2] + gr[3] * dw[3];
        zh[idx] = acc;
        txr[1 + h] = f2bf(acc);
    } else if (tid < 159) {
        txr[tid] = f2bf(sig[(size_t)b * 30 + (tid - 129)]);
    } else {
        txr[159] = 0;
    }
}

// ---------------------------------------------------------------------------
// u2: z1 = z + 0.5(f+fn)dt + 0.5*((g+gn).dW); fused readout y = z1.Wr + c0.Wr + br
// one wave per row m (4 h-values per lane), wave shuffle reduction.
// ---------------------------------------------------------------------------
__global__ __launch_bounds__(128) void u2_kernel(
    float* __restrict__ z,
    const float* __restrict__ f, const float* __restrict__ fn,
    const float* __restrict__ g, const float* __restrict__ gn,
    const float* __restrict__ dW, const float* __restrict__ Wr,
    const float* __restrict__ br, const float* __restrict__ c0dot,
    float* __restrict__ out, int s, float dt) {
    int m = blockIdx.x * 4 + (threadIdx.x >> 5);
    int lane = threadIdx.x & 31;
    const float* dw = dW + (size_t)m * 4;
    float part = 0.0f;
#pragma unroll
    for (int t = 0; t < 4; ++t) {
        int h = lane * 4 + t;
        size_t idx = (size_t)m * 128 + h;
        size_t gi = (size_t)m * 512 + (size_t)h * 4;
        float gsum = (g[gi + 0] + gn[gi + 0]) * dw[0] + (g[gi + 1] + gn[gi + 1]) * dw[1] +
                     (g[gi + 2] + gn[gi + 2]) * dw[2] + (g[gi + 3] + gn[gi + 3]) * dw[3];
        float z1 = z[idx] + 0.5f * (f[idx] + fn[idx]) * dt + 0.5f * gsum;
        z[idx] = z1;
        part += z1 * Wr[h];
    }
    for (int o = 16; o >= 1; o >>= 1) part += __shfl_xor(part, o, 32);
    if (lane == 0)
        out[((size_t)m * 128 + s) * 2 + 1] = part + c0dot[m >> 5] + br[0];
}

// Readout for step 0 (h0).
__global__ __launch_bounds__(128) void y0_kernel(
    const float* __restrict__ z, const float* __restrict__ Wr,
    const float* __restrict__ br, const float* __restrict__ c0dot,
    float* __restrict__ out) {
    int m = blockIdx.x * 4 + (threadIdx.x >> 5);
    int lane = threadIdx.x & 31;
    float part = 0.0f;
#pragma unroll
    for (int t = 0; t < 4; ++t) {
        int h = lane * 4 + t;
        part += z[(size_t)m * 128 + h] * Wr[h];
    }
    for (int o = 16; o >= 1; o >>= 1) part += __shfl_xor(part, o, 32);
    if (lane == 0) out[(size_t)m * 128 * 2 + 1] = part + c0dot[m >> 5] + br[0];
}

// Fill channel 0 of the output with ts.
__global__ void tchan_kernel(const float* __restrict__ ts, float* __restrict__ out) {
    int tid = blockIdx.x * blockDim.x + threadIdx.x;
    if (tid >= 1024 * 128) return;
    int l = tid & 127;
    out[(size_t)tid * 2] = ts[l];
}

// ---------------------------------------------------------------------------

extern "C" void kernel_launch(void* const* d_in, const int* in_sizes, int n_in,
                              void* d_out, int out_size, void* d_ws, size_t ws_size,
                              hipStream_t stream) {
    (void)in_sizes; (void)n_in; (void)out_size; (void)ws_size;
    const float* ts    = (const float*)d_in[0];
    const float* paths = (const float*)d_in[1];
    const float* dWall = (const float*)d_in[2];
    const float* Wi    = (const float*)d_in[3];
    const float* bi    = (const float*)d_in[4];
    const float* fW1   = (const float*)d_in[5];
    const float* fb1   = (const float*)d_in[6];
    const float* fW2   = (const float*)d_in[7];
    const float* fb2   = (const float*)d_in[8];
    const float* fW3   = (const float*)d_in[9];
    const float* fb3   = (const float*)d_in[10];
    const float* gW1   = (const float*)d_in[11];
    const float* gb1   = (const float*)d_in[12];
    const float* gW2   = (const float*)d_in[13];
    const float* gb2   = (const float*)d_in[14];
    const float* gW3   = (const float*)d_in[15];
    const float* gb3   = (const float*)d_in[16];
    const float* Wr    = (const float*)d_in[17];
    const float* br    = (const float*)d_in[18];
    float* out = (float*)d_out;

    // ---- workspace layout (256B aligned regions) ----
    char* base = (char*)d_ws;
    size_t off = 0;
    auto alloc = [&](size_t bytes) {
        char* p = base + off;
        off = (off + bytes + 255) & ~(size_t)255;
        return p;
    };
    float*          sig    = (float*)alloc(32 * 30 * 4);
    float*          c0dot  = (float*)alloc(32 * 4);
    unsigned short* tx     = (unsigned short*)alloc(1024 * 160 * 2);
    unsigned short* act1   = (unsigned short*)alloc((size_t)1024 * 512 * 2);
    unsigned short* act2   = (unsigned short*)alloc((size_t)1024 * 512 * 2);
    float*          z      = (float*)alloc((size_t)1024 * 128 * 4);
    float*          zh     = (float*)alloc((size_t)1024 * 128 * 4);
    float*          fA     = (float*)alloc((size_t)1024 * 128 * 4);
    float*          fB     = (float*)alloc((size_t)1024 * 128 * 4);
    float*          gA     = (float*)alloc((size_t)1024 * 512 * 4);
    float*          gB     = (float*)alloc((size_t)1024 * 512 * 4);
    unsigned int*   W1pack = (unsigned int*)alloc((size_t)5 * 32 * 256 * 4);  // Kp=160, N=512
    unsigned int*   W2pack = (unsigned int*)alloc((size_t)8 * 32 * 256 * 4);  // K=256, N=512
    unsigned int*   W3pack = (unsigned int*)alloc((size_t)8 * 40 * 256 * 4);  // K=256, N=640

    const float dt = 1.0f / 127.0f;

    // ---- one-time prep ----
    sig_kernel<<<1, 32, 0, stream>>>(paths, sig);
    c0dot_kernel<<<1, 32, 0, stream>>>(sig, Wr, c0dot);

    auto packLaunch = [&](const float* W, int K, int N, unsigned int* dst, int kT, int ntOff) {
        int total = kT * (N >> 4) * 256;
        pack_b_kernel<<<(total + 255) / 256, 256, 0, stream>>>(W, K, N, dst, kT, ntOff);
    };
    packLaunch(fW1, 159, 256, W1pack, 5, 0);
    packLaunch(gW1, 159, 256, W1pack, 5, 16);
    packLaunch(fW2, 256, 256, W2pack, 8, 0);
    packLaunch(gW2, 256, 256, W2pack, 8, 16);
    packLaunch(fW3, 256, 128, W3pack, 8, 0);
    packLaunch(gW3, 256, 512, W3pack, 8, 8);

    tchan_kernel<<<512, 256, 0, stream>>>(ts, out);

    h0_kernel<<<1024, 160, 0, stream>>>(paths, sig, Wi, bi, ts, z, zh, tx);
    y0_kernel<<<256, 128, 0, stream>>>(z, Wr, br, c0dot, out);

    auto mlp_eval = [&](float* fOut, float* gOut) {
        // layer1: tx[1024,160] -> act1[1024,512] (drift | diffusion halves)
        gemm_lipswish<5><<<128, 128, 0, stream>>>(tx, 160, 0, (const uint4*)W1pack, 32,
                                                  fb1, gb1, 256, act1, 512);
        // layer2: act1 -> act2
        gemm_lipswish<8><<<128, 128, 0, stream>>>(act1, 512, 256, (const uint4*)W2pack, 32,
                                                  fb2, gb2, 256, act2, 512);
        // layer3: act2 -> f[1024,128], g[1024,512]
        gemm_l3_tanh<8><<<160, 128, 0, stream>>>(act2, 512, 256, (const uint4*)W3pack, 40,
                                                 fb3, gb3, 128, fOut, gOut);
    };

    // f0, g0 at t = ts[0]
    mlp_eval(fA, gA);

    float* fc = fA; float* gc = gA;
    float* fn = fB; float* gn = gB;
    for (int s = 1; s < 128; ++s) {
        const float* dW = dWall + (size_t)(s - 1) * 1024 * 4;
        u1_kernel<<<1024, 160, 0, stream>>>(z, zh, fc, gc, dW, sig, ts, s, dt, tx);
        mlp_eval(fn, gn);
        u2_kernel<<<256, 128, 0, stream>>>(z, fc, fn, gc, gn, dW, Wr, br, c0dot, out, s, dt);
        std::swap(fc, fn);
        std::swap(gc, gn);
    }
}
